// GaussianFeatureLeverage_7584912245063
// MI455X (gfx1250) — matrix operand order repaired and checked by reading, not compiled, not run on hardware
//
#include <hip/hip_runtime.h>
#include <hip/hip_bf16.h>

typedef __attribute__((ext_vector_type(16))) _Float16 v16h;
typedef __attribute__((ext_vector_type(8)))  _Float16 v8h;
typedef __attribute__((ext_vector_type(8)))  float    v8f;

#define HH    48
#define WWID  160
#define HWPIX (HH * WWID)   // 7680 pixels == 7680 gaussians
#define NG    HWPIX
#define NSEG  16
#define GSEG  (NG / NSEG)   // 480 gaussians per segment
#define CSEG  (GSEG / 32)   // 15 chunks of 32 per segment

// ---------------------------------------------------------------- activations
__device__ __forceinline__ float gelu_tanh(float x) {
    float x3 = x * x * x;
    float t  = tanhf(0.7978845608028654f * (x + 0.044715f * x3));
    return 0.5f * x * (1.0f + t);
}
__device__ __forceinline__ float eluf(float x) {
    return x > 0.0f ? x : expm1f(x);
}
__device__ __forceinline__ float alpha_of(float u, float v, float A, float Bc,
                                          float Cc, float op, float pxf, float pyf) {
    float dx = pxf - u, dy = pyf - v;
    float power = -0.5f * (A * dx * dx + Cc * dy * dy) - Bc * dx * dy;
    float alpha = 0.0f;
    if (power <= 0.0f)
        alpha = fminf(fmaxf(op * __expf(power), 0.0f), 0.99f);
    return alpha;
}

// ------------------------------------------------------- weight f32 -> f16 pad
// Layout: wf[CoutPad][Kpad], K reordered tap-major: k = tap*Cin + ci
__global__ void cvt_weights(const float* __restrict__ w, _Float16* __restrict__ wf,
                            int Cout, int Cin, int Kpad, int CoutPad) {
    int i = blockIdx.x * blockDim.x + threadIdx.x;
    int total = CoutPad * Kpad;
    if (i >= total) return;
    int K   = Cin * 9;
    int row = i / Kpad;
    int k   = i - row * Kpad;
    float v = 0.0f;
    if (row < Cout && k < K) {
        int tap = k / Cin;
        int ci  = k - tap * Cin;
        v = w[(size_t)row * K + ci * 9 + tap];
    }
    wf[i] = (_Float16)v;
}

// --------------------------------------------------------- depth + unprojection
__global__ void depth_pos(const float* __restrict__ disp,
                          const float* __restrict__ invK, // 4x4 row-major
                          float* __restrict__ pos) {      // [NG][3]
    int n = blockIdx.x * blockDim.x + threadIdx.x;
    if (n >= HWPIX) return;
    float d      = disp[n];
    float scaled = (1.0f / 100.0f) + (1.0f / 0.1f - 1.0f / 100.0f) * d;
    float depth  = 1.0f / scaled;
    depth = fminf(fmaxf(depth, 0.1f), 100.0f);
    float gx = (float)(n % WWID), gy = (float)(n / WWID);
    float X = (invK[0] * gx + invK[1] * gy + invK[2])  * depth;
    float Y = (invK[4] * gx + invK[5] * gy + invK[6])  * depth;
    float Z = (invK[8] * gx + invK[9] * gy + invK[10]) * depth;
    pos[n * 3 + 0] = X; pos[n * 3 + 1] = Y; pos[n * 3 + 2] = Z;
}

// --------------------------------------------------------------- stable argsort
__global__ void rank_sort(const float* __restrict__ pos, int* __restrict__ order) {
    int i = blockIdx.x * blockDim.x + threadIdx.x;
    if (i >= NG) return;
    float zi = pos[i * 3 + 2];
    int r = 0;
    for (int j = 0; j < NG; ++j) {
        float zj = pos[j * 3 + 2];
        r += (zj < zi) || (zj == zi && j < i);
    }
    order[r] = i;
}

// -------------------------------------------- per-gaussian conic prep + gather
__global__ void gauss_prep(const int* __restrict__ order,
                           const float* __restrict__ pos,     // [NG][3]
                           const float* __restrict__ rotmap,  // [4][HW]
                           const float* __restrict__ sclmap,  // [3][HW]
                           const float* __restrict__ opamap,  // [1][HW]
                           const float* __restrict__ feat,    // [64][HW]
                           const float* __restrict__ Kmat,    // 4x4
                           float* __restrict__ params,        // [NG][6] sorted
                           _Float16* __restrict__ featT) {    // [64][NG] sorted f16
    int s = blockIdx.x * blockDim.x + threadIdx.x;
    if (s >= NG) return;
    int g = order[s];
    float qw = rotmap[0 * HWPIX + g], qx = rotmap[1 * HWPIX + g];
    float qy = rotmap[2 * HWPIX + g], qz = rotmap[3 * HWPIX + g];
    float nrm = fmaxf(sqrtf(qw * qw + qx * qx + qy * qy + qz * qz), 1e-12f);
    qw /= nrm; qx /= nrm; qy /= nrm; qz /= nrm;
    float sc[3] = { fabsf(sclmap[0 * HWPIX + g]),
                    fabsf(sclmap[1 * HWPIX + g]),
                    fabsf(sclmap[2 * HWPIX + g]) };
    float opa = 1.0f / (1.0f + expf(-opamap[g]));
    float px = pos[g * 3 + 0], py = pos[g * 3 + 1], pz = pos[g * 3 + 2];
    float fx = Kmat[0], fy = Kmat[5], cx = Kmat[2], cy = Kmat[6];
    float u = fx * px / pz + cx;
    float v = fy * py / pz + cy;
    float R[3][3] = {
        {1 - 2 * (qy * qy + qz * qz), 2 * (qx * qy - qw * qz), 2 * (qx * qz + qw * qy)},
        {2 * (qx * qy + qw * qz), 1 - 2 * (qx * qx + qz * qz), 2 * (qy * qz - qw * qx)},
        {2 * (qx * qz - qw * qy), 2 * (qy * qz + qw * qx), 1 - 2 * (qx * qx + qy * qy)}};
    float M[3][3], Sig[3][3];
    #pragma unroll
    for (int i = 0; i < 3; ++i)
        #pragma unroll
        for (int j = 0; j < 3; ++j) M[i][j] = R[i][j] * sc[j];
    #pragma unroll
    for (int i = 0; i < 3; ++i)
        #pragma unroll
        for (int k = 0; k < 3; ++k)
            Sig[i][k] = M[i][0] * M[k][0] + M[i][1] * M[k][1] + M[i][2] * M[k][2];
    float iz = 1.0f / pz;
    float J0[3] = { fx * iz, 0.0f,    -fx * px * iz * iz };
    float J1[3] = { 0.0f,    fy * iz, -fy * py * iz * iz };
    float T0[3], T1[3];
    #pragma unroll
    for (int k = 0; k < 3; ++k) {
        T0[k] = J0[0] * Sig[0][k] + J0[1] * Sig[1][k] + J0[2] * Sig[2][k];
        T1[k] = J1[0] * Sig[0][k] + J1[1] * Sig[1][k] + J1[2] * Sig[2][k];
    }
    float cov00 = T0[0] * J0[0] + T0[1] * J0[1] + T0[2] * J0[2];
    float cov01 = T0[0] * J1[0] + T0[1] * J1[1] + T0[2] * J1[2];
    float cov11 = T1[0] * J1[0] + T1[1] * J1[1] + T1[2] * J1[2];
    float a = cov00 + 0.3f, bq = cov01, c = cov11 + 0.3f;
    float det = a * c - bq * bq;
    params[s * 6 + 0] = u;
    params[s * 6 + 1] = v;
    params[s * 6 + 2] = c / det;    // A
    params[s * 6 + 3] = -bq / det;  // Bc
    params[s * 6 + 4] = a / det;    // C
    params[s * 6 + 5] = opa;
    #pragma unroll 8
    for (int ch = 0; ch < 64; ++ch)
        featT[(size_t)ch * NG + s] = (_Float16)feat[(size_t)ch * HWPIX + g];
}

// ----------------------------------------------- 3x3 conv as implicit WMMA GEMM
// One wave32 per 16-pixel tile (tiles never cross rows: 160 % 16 == 0).
// K reordered tap-major (k = tap*CIN + ci) so per-chunk tap is uniform for
// CIN in {64,128}; CIN is a template arg so div/mod folds to constants.
// A layout (16-bit 16x32): lane l -> row l&15; lanes 0-15 K0..15,
// lanes 16-31 K16..31.  B layout (32x16): lanes 0-15 K0..15,
// lanes 16-31 K16..31, contiguous -> one 32B load.
template <int CIN, int NTILES, int PAD, int ACT>
__global__ __launch_bounds__(32) void conv3x3_wmma(
    const float* __restrict__ in,
    const _Float16* __restrict__ wf16, int Kpad,
    const float* __restrict__ bias, int Cout,
    float* __restrict__ out) {
    const int lane = threadIdx.x;
    const int M0   = blockIdx.x * 16;
    const int K    = CIN * 9;
    const int m    = M0 + (lane & 15);
    const int py   = m / WWID, px = m % WWID;
    const bool hi  = lane >= 16;

    v8f acc[NTILES] = {};

    for (int kbase = 0; kbase < Kpad; kbase += 32) {
        v16h a;
        #pragma unroll
        for (int j = 0; j < 16; ++j) {
            int kl = j + (hi ? 16 : 0);
            int k  = kbase + kl;
            float v = 0.0f;
            if (k < K) {
                int tap = k / CIN;            // compile-time divisor
                int ci  = k - tap * CIN;
                int dy  = tap / 3 - 1;
                int dx  = tap - (dy + 1) * 3 - 1;
                int yy = py + dy, xx = px + dx;
                if (PAD == 0) { // edge
                    yy = yy < 0 ? 0 : (yy >= HH ? HH - 1 : yy);
                    xx = xx < 0 ? 0 : (xx >= WWID ? WWID - 1 : xx);
                } else {        // reflect
                    yy = yy < 0 ? -yy : (yy >= HH ? 2 * HH - 2 - yy : yy);
                    xx = xx < 0 ? -xx : (xx >= WWID ? 2 * WWID - 2 - xx : xx);
                }
                v = in[(size_t)ci * HWPIX + yy * WWID + xx];
            }
            a[j] = (_Float16)v;
        }
        #pragma unroll
        for (int t = 0; t < NTILES; ++t) {
            int row = t * 16 + (lane & 15);
            const _Float16* wrow = wf16 + (size_t)row * Kpad + kbase + (hi ? 16 : 0);
            v16h b = *(const v16h*)wrow; // 32B aligned
            acc[t] = __builtin_amdgcn_wmma_f32_16x16x32_f16(
                false, a, false, b, (short)0, acc[t], false, false);
        }
    }
    // C layout: lane n(0-15): vgpr r -> (M=r, N=n); lanes 16-31: M=r+8.
    #pragma unroll
    for (int t = 0; t < NTILES; ++t) {
        int ch = t * 16 + (lane & 15);
        if (ch < Cout) {
            float bv = bias[ch];
            #pragma unroll
            for (int r = 0; r < 8; ++r) {
                int pix = M0 + r + (hi ? 8 : 0);
                float v = acc[t][r] + bv;
                if (ACT == 1)      v = gelu_tanh(v);
                else if (ACT == 2) v = eluf(v);
                out[(size_t)ch * HWPIX + pix] = v;
            }
        }
    }
}

// --------------------------------- pass 1: per-(segment,pixel) sum log1p(-alpha)
// id = seg*NG + pixel -> waves share a segment -> uniform param loads.
__global__ void seg_logsum(const float* __restrict__ params,
                           float* __restrict__ segsum) {     // [NSEG][NG]
    int id = blockIdx.x * blockDim.x + threadIdx.x;
    if (id >= NSEG * NG) return;
    int seg = id / NG, pix = id - seg * NG;
    float pxf = (float)(pix % WWID), pyf = (float)(pix / WWID);
    const float* pr = params + (size_t)seg * GSEG * 6;
    float s = 0.0f;
    for (int g = 0; g < GSEG; ++g) {
        float alpha = alpha_of(pr[g * 6 + 0], pr[g * 6 + 1], pr[g * 6 + 2],
                               pr[g * 6 + 3], pr[g * 6 + 4], pr[g * 6 + 5],
                               pxf, pyf);
        s += __logf(1.0f - alpha);
    }
    segsum[id] = s;
}

// ------------------------------- pass 2: exclusive scan of segment sums / pixel
__global__ void seg_scan(const float* __restrict__ segsum,
                         float* __restrict__ segoff) {       // [NSEG][NG]
    int pix = blockIdx.x * blockDim.x + threadIdx.x;
    if (pix >= NG) return;
    float run = 0.0f;
    #pragma unroll
    for (int s = 0; s < NSEG; ++s) {
        segoff[(size_t)s * NG + pix] = run;
        run += segsum[(size_t)s * NG + pix];
    }
}

// ---------------------------------------------------- composited WMMA rasterize
// grid.x = pixel tile (480), grid.y = segment (NSEG or 1).
// seg_mode=1: logT seeded from segoff, output to per-segment partial buffer.
// seg_mode=0: single pass over all chunks, output directly to lev.
__global__ __launch_bounds__(32) void rasterize_wmma(
    const float* __restrict__ params,    // [NG][6] sorted
    const _Float16* __restrict__ featT,  // [64][NG] sorted f16
    const float* __restrict__ segoff,    // [NSEG][NG] (seg_mode only)
    float* __restrict__ dst,             // lev or lev_part
    int chunks, int seg_mode) {
    __shared__ __align__(16) float    sp[32 * 6];
    __shared__ __align__(16) _Float16 swgt[16 * 32];

    const int  lane = threadIdx.x;
    const int  P0   = blockIdx.x * 16;
    const int  seg  = blockIdx.y;
    const int  c0   = seg * chunks;
    const bool hi   = lane >= 16;
    const int  lp   = lane & 15;

    v8f acc[4] = {};
    float logT = seg_mode ? segoff[(size_t)seg * NG + P0 + lp] : 0.0f;
    float* outp = dst + (seg_mode ? (size_t)seg * 64 * NG : 0);
    const float pxf = (float)((P0 + lp) % WWID);
    const float pyf = (float)((P0 + lp) / WWID);

    for (int c = c0; c < c0 + chunks; ++c) {
        __syncthreads(); // previous chunk's LDS reads done
        {   // stage params row per lane
            const float* pr = params + (size_t)(c * 32 + lane) * 6;
            #pragma unroll
            for (int q = 0; q < 6; ++q) sp[lane * 6 + q] = pr[q];
            if (c + 1 < c0 + chunks)
                __builtin_prefetch(params + (size_t)((c + 1) * 32 + lane) * 6, 0, 0);
        }
        __syncthreads();
        if (!hi) { // sequential transmittance scan; weights stay in 2 v16h regs
            v16h w0, w1;
            #pragma unroll
            for (int g = 0; g < 32; ++g) {
                float alpha = alpha_of(sp[g * 6 + 0], sp[g * 6 + 1], sp[g * 6 + 2],
                                       sp[g * 6 + 3], sp[g * 6 + 4], sp[g * 6 + 5],
                                       pxf, pyf);
                float wgt = alpha * __expf(logT);
                logT += __logf(1.0f - alpha);
                if (g < 16) w0[g] = (_Float16)wgt;
                else        w1[g - 16] = (_Float16)wgt;
            }
            *(v16h*)(swgt + lp * 32)      = w0;
            *(v16h*)(swgt + lp * 32 + 16) = w1;
        }
        __syncthreads();
        // A fragment: two aligned 16B LDS loads + shuffle
        const _Float16* rowp = swgt + lp * 32 + (hi ? 16 : 0);
        v8h r0 = *(const v8h*)(rowp);
        v8h r1 = *(const v8h*)(rowp + 8);
        v16h a = __builtin_shufflevector(r0, r1, 0, 1, 2, 3, 4, 5, 6, 7,
                                         8, 9, 10, 11, 12, 13, 14, 15);
        // B fragments: one aligned 32B global load each (featT is [ch][g])
        const int gbase = c * 32 + (hi ? 16 : 0);
        #pragma unroll
        for (int t = 0; t < 4; ++t) {
            v16h b = *(const v16h*)(featT + (size_t)(t * 16 + lp) * NG + gbase);
            acc[t] = __builtin_amdgcn_wmma_f32_16x16x32_f16(
                false, a, false, b, (short)0, acc[t], false, false);
        }
    }
    #pragma unroll
    for (int t = 0; t < 4; ++t) {
        int ch = t * 16 + lp;
        #pragma unroll
        for (int r = 0; r < 8; ++r) {
            int pix = P0 + r + (hi ? 8 : 0);
            outp[(size_t)ch * HWPIX + pix] = acc[t][r];
        }
    }
}

// ------------------------------------ deterministic fixed-order partial reduce
__global__ void seg_reduce(const float* __restrict__ part,  // [NSEG][64][NG]
                           float* __restrict__ lev) {       // [64][NG]
    int i = blockIdx.x * blockDim.x + threadIdx.x;
    if (i >= 64 * NG) return;
    float s = 0.0f;
    #pragma unroll
    for (int sg = 0; sg < NSEG; ++sg)
        s += part[(size_t)sg * 64 * NG + i];
    lev[i] = s;
}

// ----------------------------------------------------------------------- launch
extern "C" void kernel_launch(void* const* d_in, const int* in_sizes, int n_in,
                              void* d_out, int out_size, void* d_ws, size_t ws_size,
                              hipStream_t stream) {
    const float* init_feature = (const float*)d_in[0];
    const float* disp_init    = (const float*)d_in[1];
    const float* inv_K        = (const float*)d_in[2];
    const float* Kmat         = (const float*)d_in[3];
    const float* rot_w1 = (const float*)d_in[4];  const float* rot_b1 = (const float*)d_in[5];
    const float* rot_w2 = (const float*)d_in[6];  const float* rot_b2 = (const float*)d_in[7];
    const float* scl_w1 = (const float*)d_in[8];  const float* scl_b1 = (const float*)d_in[9];
    const float* scl_w2 = (const float*)d_in[10]; const float* scl_b2 = (const float*)d_in[11];
    const float* opa_w1 = (const float*)d_in[12]; const float* opa_b1 = (const float*)d_in[13];
    const float* opa_w2 = (const float*)d_in[14]; const float* opa_b2 = (const float*)d_in[15];
    const float* fl_w1  = (const float*)d_in[16]; const float* fl_b1  = (const float*)d_in[17];
    const float* fl_w2  = (const float*)d_in[18]; const float* fl_b2  = (const float*)d_in[19];
    const float* fr_w   = (const float*)d_in[20]; const float* fr_b   = (const float*)d_in[21];
    float* out = (float*)d_out;

    char* base = (char*)d_ws;
    size_t off = 0;
    auto take = [&](size_t bytes) -> void* {
        void* p = base + off;
        off = (off + bytes + 255) & ~(size_t)255;
        return p;
    };
    _Float16* wf_rot1 = (_Float16*)take(16  * 1152 * 2);
    _Float16* wf_rot2 = (_Float16*)take(16  * 96   * 2);
    _Float16* wf_scl1 = (_Float16*)take(16  * 1152 * 2);
    _Float16* wf_scl2 = (_Float16*)take(16  * 64   * 2);
    _Float16* wf_opa1 = (_Float16*)take(16  * 1152 * 2);
    _Float16* wf_opa2 = (_Float16*)take(16  * 32   * 2);
    _Float16* wf_fl1  = (_Float16*)take(128 * 1152 * 2);
    _Float16* wf_fl2  = (_Float16*)take(64  * 1152 * 2);
    _Float16* wf_fr   = (_Float16*)take(128 * 576  * 2);
    float* act_rot1 = (float*)take((size_t)8   * HWPIX * 4);
    float* act_rot2 = (float*)take((size_t)4   * HWPIX * 4);
    float* act_scl1 = (float*)take((size_t)6   * HWPIX * 4);
    float* act_scl2 = (float*)take((size_t)3   * HWPIX * 4);
    float* act_opa1 = (float*)take((size_t)2   * HWPIX * 4);
    float* act_opa2 = (float*)take((size_t)1   * HWPIX * 4);
    float* act_fl1  = (float*)take((size_t)128 * HWPIX * 4);
    float* act_fl2  = (float*)take((size_t)64  * HWPIX * 4);
    float* pos      = (float*)take((size_t)NG * 3 * 4);
    int*   order    = (int*)  take((size_t)NG * 4);
    float* gparams  = (float*)take((size_t)NG * 6 * 4);
    _Float16* featT = (_Float16*)take((size_t)64 * NG * 2);
    float* lev      = (float*)take((size_t)64 * HWPIX * 4);
    // segmented-rasterize scratch (optional, ~32.5 MB)
    float* segsum   = (float*)take((size_t)NSEG * NG * 4);
    float* segoff   = (float*)take((size_t)NSEG * NG * 4);
    float* lev_part = (float*)take((size_t)NSEG * 64 * NG * 4);
    const bool use_seg = (off <= ws_size);   // constant for a given harness setup

    auto cvt = [&](const float* w, _Float16* wf, int Cout, int Cin, int Kpad, int CoutPad) {
        int total = CoutPad * Kpad;
        cvt_weights<<<(total + 255) / 256, 256, 0, stream>>>(w, wf, Cout, Cin, Kpad, CoutPad);
    };
    cvt(rot_w1, wf_rot1, 8,   128, 1152, 16);
    cvt(rot_w2, wf_rot2, 4,   8,   96,   16);
    cvt(scl_w1, wf_scl1, 6,   128, 1152, 16);
    cvt(scl_w2, wf_scl2, 3,   6,   64,   16);
    cvt(opa_w1, wf_opa1, 2,   128, 1152, 16);
    cvt(opa_w2, wf_opa2, 1,   2,   32,   16);
    cvt(fl_w1,  wf_fl1,  128, 128, 1152, 128);
    cvt(fl_w2,  wf_fl2,  64,  128, 1152, 64);
    cvt(fr_w,   wf_fr,   128, 64,  576,  128);

    depth_pos<<<(HWPIX + 255) / 256, 256, 0, stream>>>(disp_init, inv_K, pos);

    const dim3 cgrid(HWPIX / 16), cblk(32);
    // heads: conv -> gelu -> conv (post ops in gauss_prep)
    conv3x3_wmma<128, 1, 0, 1><<<cgrid, cblk, 0, stream>>>(init_feature, wf_rot1, 1152, rot_b1, 8, act_rot1);
    conv3x3_wmma<8,   1, 0, 0><<<cgrid, cblk, 0, stream>>>(act_rot1,     wf_rot2, 96,   rot_b2, 4, act_rot2);
    conv3x3_wmma<128, 1, 0, 1><<<cgrid, cblk, 0, stream>>>(init_feature, wf_scl1, 1152, scl_b1, 6, act_scl1);
    conv3x3_wmma<6,   1, 0, 0><<<cgrid, cblk, 0, stream>>>(act_scl1,     wf_scl2, 64,   scl_b2, 3, act_scl2);
    conv3x3_wmma<128, 1, 0, 1><<<cgrid, cblk, 0, stream>>>(init_feature, wf_opa1, 1152, opa_b1, 2, act_opa1);
    conv3x3_wmma<2,   1, 0, 0><<<cgrid, cblk, 0, stream>>>(act_opa1,     wf_opa2, 32,   opa_b2, 1, act_opa2);
    // feature path: elu(conv(reflect)) x2
    conv3x3_wmma<128, 8, 1, 2><<<cgrid, cblk, 0, stream>>>(init_feature, wf_fl1, 1152, fl_b1, 128, act_fl1);
    conv3x3_wmma<128, 4, 1, 2><<<cgrid, cblk, 0, stream>>>(act_fl1,      wf_fl2, 1152, fl_b2, 64,  act_fl2);

    rank_sort<<<(NG + 255) / 256, 256, 0, stream>>>(pos, order);
    gauss_prep<<<(NG + 255) / 256, 256, 0, stream>>>(order, pos, act_rot2, act_scl2,
                                                     act_opa2, act_fl2, Kmat, gparams, featT);

    if (use_seg) {
        seg_logsum<<<(NSEG * NG + 255) / 256, 256, 0, stream>>>(gparams, segsum);
        seg_scan<<<(NG + 255) / 256, 256, 0, stream>>>(segsum, segoff);
        rasterize_wmma<<<dim3(HWPIX / 16, NSEG), dim3(32), 0, stream>>>(
            gparams, featT, segoff, lev_part, CSEG, 1);
        seg_reduce<<<(64 * NG + 255) / 256, 256, 0, stream>>>(lev_part, lev);
    } else {
        rasterize_wmma<<<dim3(HWPIX / 16, 1), dim3(32), 0, stream>>>(
            gparams, featT, lev /*unused*/, lev, NG / 32, 0);
    }

    // final convblock: elu(conv3x3(lev, reflect)) -> d_out (1,128,48,160)
    conv3x3_wmma<64, 8, 1, 2><<<cgrid, cblk, 0, stream>>>(lev, wf_fr, 576, fr_b, 128, out);
}
